// NeRF_Complete_13967233646654
// MI455X (gfx1250) — matrix operand order repaired and checked by reading, not compiled, not run on hardware
//
#include <hip/hip_runtime.h>
#include <hip/hip_bf16.h>

typedef __attribute__((ext_vector_type(16))) _Float16 v16h;
typedef __attribute__((ext_vector_type(8)))  float    v8f;

#define NB 1024
#define NC 64
#define NF 128
#define NT 192          // NC + NF
#define LDACT 320       // activation row stride (256 hidden + 64 concat slot)
#define PI_F 3.14159265358979f

// ---------------------------------------------------------------------------
// Weight f32 [K,N] -> f16 TRANSPOSED [N,Kp] with zero K-padding.
// ---------------------------------------------------------------------------
__global__ void k_pad_weight_t(const float* __restrict__ w, _Float16* __restrict__ dst,
                               int K, int Kp, int N) {
  int i = blockIdx.x * 256 + threadIdx.x;
  if (i >= Kp * N) return;
  int n = i / Kp, k = i - n * Kp;
  float v = (k < K) ? w[(size_t)k * N + n] : 0.0f;
  dst[i] = (_Float16)v;
}

// ---------------------------------------------------------------------------
// Coarse sample prep: stratified z, points, positional encoding (63 -> pad 64)
// ---------------------------------------------------------------------------
__global__ void k_prep_coarse(const float* __restrict__ ro, const float* __restrict__ rd,
                              const float* __restrict__ u,
                              const int* __restrict__ nearp, const int* __restrict__ farp,
                              _Float16* __restrict__ pin, _Float16* __restrict__ actA,
                              _Float16* __restrict__ actB, float* __restrict__ zc) {
  int i = blockIdx.x * 256 + threadIdx.x;
  if (i >= NB * NC) return;
  int b = i / NC, s = i - b * NC;
  float nearf = (float)nearp[0], farf = (float)farp[0];
  float t = (float)s / (float)NC + (1.0f / (float)NC) * u[i];
  float z = nearf * (1.0f - t) + farf * t;
  zc[i] = z;
  float p[3];
  #pragma unroll
  for (int c = 0; c < 3; ++c) p[c] = ro[b * 3 + c] + rd[b * 3 + c] * z;
  _Float16 e[64];
  #pragma unroll
  for (int c = 0; c < 3; ++c) e[c] = (_Float16)p[c];
  for (int L = 0; L < 10; ++L) {
    float f = (float)(1 << L) * PI_F;
    #pragma unroll
    for (int c = 0; c < 3; ++c) {
      float a = p[c] * f;
      e[3 + L * 6 + c]     = (_Float16)__sinf(a);
      e[3 + L * 6 + 3 + c] = (_Float16)__cosf(a);
    }
  }
  e[63] = (_Float16)0.0f;
  _Float16* pp = pin + (size_t)i * 64;
  _Float16* pa = actA + (size_t)i * LDACT + 256;
  _Float16* pb = actB + (size_t)i * LDACT + 256;
  #pragma unroll
  for (int c = 0; c < 8; ++c) {
    uint4 v = *(const uint4*)&e[c * 8];
    *(uint4*)&pp[c * 8] = v;
    *(uint4*)&pa[c * 8] = v;
    *(uint4*)&pb[c * 8] = v;
  }
}

// Fine sample prep: z comes from the sorted z_all buffer
__global__ void k_prep_fine(const float* __restrict__ ro, const float* __restrict__ rd,
                            const float* __restrict__ zall,
                            _Float16* __restrict__ pin, _Float16* __restrict__ actA,
                            _Float16* __restrict__ actB) {
  int i = blockIdx.x * 256 + threadIdx.x;
  if (i >= NB * NT) return;
  int b = i / NT;
  float z = zall[i];
  float p[3];
  #pragma unroll
  for (int c = 0; c < 3; ++c) p[c] = ro[b * 3 + c] + rd[b * 3 + c] * z;
  _Float16 e[64];
  #pragma unroll
  for (int c = 0; c < 3; ++c) e[c] = (_Float16)p[c];
  for (int L = 0; L < 10; ++L) {
    float f = (float)(1 << L) * PI_F;
    #pragma unroll
    for (int c = 0; c < 3; ++c) {
      float a = p[c] * f;
      e[3 + L * 6 + c]     = (_Float16)__sinf(a);
      e[3 + L * 6 + 3 + c] = (_Float16)__cosf(a);
    }
  }
  e[63] = (_Float16)0.0f;
  _Float16* pp = pin + (size_t)i * 64;
  _Float16* pa = actA + (size_t)i * LDACT + 256;
  _Float16* pb = actB + (size_t)i * LDACT + 256;
  #pragma unroll
  for (int c = 0; c < 8; ++c) {
    uint4 v = *(const uint4*)&e[c * 8];
    *(uint4*)&pp[c * 8] = v;
    *(uint4*)&pa[c * 8] = v;
    *(uint4*)&pb[c * 8] = v;
  }
}

// Direction encoding (27 -> pad 32) into cols 256..287 of activation buffer
__global__ void k_din(const float* __restrict__ rd, _Float16* __restrict__ act,
                      int nS, int M) {
  int i = blockIdx.x * 256 + threadIdx.x;
  if (i >= M) return;
  int b = i / nS;
  float d[3] = { rd[b * 3 + 0], rd[b * 3 + 1], rd[b * 3 + 2] };
  _Float16 e[32];
  #pragma unroll
  for (int j = 0; j < 32; ++j) e[j] = (_Float16)0.0f;
  #pragma unroll
  for (int c = 0; c < 3; ++c) e[c] = (_Float16)d[c];
  for (int L = 0; L < 4; ++L) {
    float f = (float)(1 << L) * PI_F;
    #pragma unroll
    for (int c = 0; c < 3; ++c) {
      float a = d[c] * f;
      e[3 + L * 6 + c]     = (_Float16)__sinf(a);
      e[3 + L * 6 + 3 + c] = (_Float16)__cosf(a);
    }
  }
  _Float16* pa = act + (size_t)i * LDACT + 256;
  #pragma unroll
  for (int c = 0; c < 4; ++c) *(uint4*)&pa[c * 8] = *(const uint4*)&e[c * 8];
}

// ---------------------------------------------------------------------------
// WMMA GEMM: O[M,N] = act(A[M,K] @ W[K,N] + bias), f16 in, f32 acc, f16 out.
// Wt is the K-padded TRANSPOSED weight [N,K].
// 8 waves/block; wave owns a 16x128 strip = 8 x v_wmma_f32_16x16x32_f16/K-step.
// Double-buffered LDS, direct b128 global->LDS staging (no register staging),
// stage-next-then-compute ordering: ONE barrier per K-step.
// ---------------------------------------------------------------------------
__global__ __launch_bounds__(256) void k_gemm(
    const _Float16* __restrict__ A, const _Float16* __restrict__ Wt,
    const float* __restrict__ bias, _Float16* __restrict__ O,
    int K, int lda, int ldo, int N, int act) {
  __shared__ _Float16 sA[2][128 * 32];  // BM x 32, K-contiguous rows
  __shared__ _Float16 sW[2][256 * 32];  // N  x 32, K-contiguous cols
  int tid = threadIdx.x;
  int colG = N >> 7;                    // 2 (N=256) or 1 (N=128)
  int rowG = 8 / colG;                  // 4 or 8
  int BM = rowG << 4;                   // 64 or 128
  int rowBase = blockIdx.x * BM;
  int wave = tid >> 5, lane = tid & 31;
  int wm = wave / colG, wn = wave - wm * colG;
  int l16 = lane & 15, lHi = lane >> 4;

  // LDS offset for 16B chunk idx = (idx>>2)*32 + (idx&3)*8 = idx*8 halves
  auto stage = [&](int k0, int buf) {
    int nA = BM << 2;
    for (int idx = tid; idx < nA; idx += 256)
      *(uint4*)&sA[buf][idx * 8] =
          *(const uint4*)&A[(size_t)(rowBase + (idx >> 2)) * lda + k0 + (idx & 3) * 8];
    int nW = N << 2;
    for (int idx = tid; idx < nW; idx += 256)
      *(uint4*)&sW[buf][idx * 8] =
          *(const uint4*)&Wt[(size_t)(idx >> 2) * K + k0 + (idx & 3) * 8];
  };

  v8f acc[8];
  v8f zero = {0.f, 0.f, 0.f, 0.f, 0.f, 0.f, 0.f, 0.f};
  #pragma unroll
  for (int t = 0; t < 8; ++t) acc[t] = zero;

  stage(0, 0);
  __syncthreads();

  int pb = 0;
  for (int k0 = 0; k0 < K; k0 += 32, pb ^= 1) {
    // stage tile k+1 into the other buffer; its latency hides under other
    // waves' WMMAs (compute below never waits on freshly staged data)
    if (k0 + 32 < K) stage(k0 + 32, pb ^ 1);
    // L2 prefetch two tiles ahead (global_prefetch_b8)
    if (k0 + 64 < K) {
      __builtin_prefetch(&A[(size_t)(rowBase + (tid >> 2)) * lda + k0 + 64], 0, 1);
      __builtin_prefetch(&Wt[(size_t)(tid & (N - 1)) * K + k0 + 64], 0, 1);
    }

    // A fragment: lane l<16 -> M=l, K in {0..7,16..23}; l>=16 -> {8..15,24..31}
    v16h af;
    const _Float16* aRow = &sA[pb][(wm * 16 + l16) * 32];
    int kHi = lHi * 8;
    #pragma unroll
    for (int h = 0; h < 16; h += 2) {
      int kk = ((h & 8) ? 16 : 0) + kHi + (h & 7);
      af[h] = aRow[kk];
      af[h + 1] = aRow[kk + 1];
    }
    #pragma unroll
    for (int t = 0; t < 8; ++t) {
      // B fragment: lane%16 = N-col, lane<16 -> K {0..7,16..23}; lane>=16 -> {8..15,24..31}
      const _Float16* bCol = &sW[pb][(wn * 128 + t * 16 + l16) * 32];
      v16h bf;
      #pragma unroll
      for (int h = 0; h < 16; ++h) bf[h] = bCol[((h & 8) ? 16 : 0) + kHi + (h & 7)];
      acc[t] = __builtin_amdgcn_wmma_f32_16x16x32_f16(
          false, af, false, bf, (short)0, acc[t], false, false);
    }

    // one barrier: prior compute (reads of pb) done before pb is re-staged
    // in the next iteration; this iteration's stores to pb^1 visible to all
    __syncthreads();
  }

  // C/D layout: VGPR r -> row = r + 8*(lane>=16); lane%16 = col
  #pragma unroll
  for (int t = 0; t < 8; ++t) {
    int col = wn * 128 + t * 16 + l16;
    float bv = bias[col];
    #pragma unroll
    for (int r = 0; r < 8; ++r) {
      int row = rowBase + wm * 16 + (lHi << 3) + r;
      float v = acc[t][r] + bv;
      if (act) v = fmaxf(v, 0.0f);
      O[(size_t)row * ldo + col] = (_Float16)v;
    }
  }
}

// ---------------------------------------------------------------------------
// density_out head: sigma = relu(x @ w[256,1] + b)
// ---------------------------------------------------------------------------
__global__ void k_sigma(const _Float16* __restrict__ act, const float* __restrict__ w,
                        const float* __restrict__ b, float* __restrict__ sig, int M) {
  int i = blockIdx.x * 256 + threadIdx.x;
  if (i >= M) return;
  const _Float16* a = act + (size_t)i * LDACT;
  float s = b[0];
  for (int k = 0; k < 256; k += 8) {
    uint4 u = *(const uint4*)&a[k];
    const _Float16* h = (const _Float16*)&u;
    #pragma unroll
    for (int j = 0; j < 8; ++j) s += (float)h[j] * w[k + j];
  }
  sig[i] = fmaxf(s, 0.0f);
}

// rgb head: sigmoid(cf @ w[128,3] + b)
__global__ void k_rgb(const _Float16* __restrict__ act, const float* __restrict__ w,
                      const float* __restrict__ b, float* __restrict__ out, int M) {
  int i = blockIdx.x * 256 + threadIdx.x;
  if (i >= M) return;
  const _Float16* a = act + (size_t)i * LDACT;
  float acc0 = b[0], acc1 = b[1], acc2 = b[2];
  for (int k = 0; k < 128; k += 8) {
    uint4 u = *(const uint4*)&a[k];
    const _Float16* h = (const _Float16*)&u;
    #pragma unroll
    for (int j = 0; j < 8; ++j) {
      float x = (float)h[j];
      acc0 += x * w[(k + j) * 3 + 0];
      acc1 += x * w[(k + j) * 3 + 1];
      acc2 += x * w[(k + j) * 3 + 2];
    }
  }
  out[(size_t)i * 3 + 0] = 1.0f / (1.0f + __expf(-acc0));
  out[(size_t)i * 3 + 1] = 1.0f / (1.0f + __expf(-acc1));
  out[(size_t)i * 3 + 2] = 1.0f / (1.0f + __expf(-acc2));
}

// ---------------------------------------------------------------------------
// Volume rendering: one thread per ray
// ---------------------------------------------------------------------------
__global__ void k_volrender(const float* __restrict__ sigma, const float* __restrict__ col,
                            const float* __restrict__ z, float* __restrict__ rgbOut,
                            float* __restrict__ wOut, int nS) {
  int ray = blockIdx.x * 256 + threadIdx.x;
  if (ray >= NB) return;
  size_t base = (size_t)ray * nS;
  float T = 1.0f, r0 = 0.f, r1 = 0.f, r2 = 0.f;
  for (int s = 0; s < nS; ++s) {
    float dist = (s < nS - 1) ? (z[base + s + 1] - z[base + s]) : 1e10f;
    float a = 1.0f - __expf(-sigma[base + s] * dist);
    float w = T * a;
    r0 += w * col[(base + s) * 3 + 0];
    r1 += w * col[(base + s) * 3 + 1];
    r2 += w * col[(base + s) * 3 + 2];
    if (wOut) wOut[base + s] = w;
    T *= (1.0f - a + 1e-10f);
  }
  rgbOut[ray * 3 + 0] = r0;
  rgbOut[ray * 3 + 1] = r1;
  rgbOut[ray * 3 + 2] = r2;
}

// ---------------------------------------------------------------------------
// Hierarchical inverse-CDF sampling + insertion-sort of 64+128 z's (1 thread/ray)
// ---------------------------------------------------------------------------
__global__ void k_sample_sort(const float* __restrict__ zc, const float* __restrict__ wc,
                              const float* __restrict__ uf, float* __restrict__ zall) {
  int ray = blockIdx.x * 256 + threadIdx.x;
  if (ray >= NB) return;
  float w[NC], cdf[NC + 1], zb[NC - 1], arr[NT];
  float sum = 0.0f;
  for (int s = 0; s < NC; ++s) { w[s] = wc[(size_t)ray * NC + s] + 1e-5f; sum += w[s]; }
  float inv = 1.0f / sum;
  cdf[0] = 0.0f;
  for (int s = 0; s < NC; ++s) cdf[s + 1] = cdf[s] + w[s] * inv;
  for (int s = 0; s < NC - 1; ++s)
    zb[s] = 0.5f * (zc[(size_t)ray * NC + s] + zc[(size_t)ray * NC + s + 1]);
  for (int s = 0; s < NC; ++s) arr[s] = zc[(size_t)ray * NC + s];
  for (int j = 0; j < NF; ++j) {
    float u = uf[(size_t)ray * NF + j];
    int idx = 0;
    for (int t = 0; t <= NC; ++t) idx += (cdf[t] <= u) ? 1 : 0;   // searchsorted right
    int below = idx - 1; if (below < 0) below = 0; if (below > NC) below = NC;
    int above = idx;     if (above > NC) above = NC;
    float cb = cdf[below], ca = cdf[above];
    int nb = below > NC - 2 ? NC - 2 : below;
    int na = above > NC - 2 ? NC - 2 : above;
    float den = ca - cb; if (den < 1e-5f) den = 1.0f;
    float tt = (u - cb) / den;
    arr[NC + j] = zb[nb] + tt * (zb[na] - zb[nb]);
  }
  for (int a = 1; a < NT; ++a) {       // insertion sort 192 elements
    float v = arr[a];
    int b = a - 1;
    while (b >= 0 && arr[b] > v) { arr[b + 1] = arr[b]; --b; }
    arr[b + 1] = v;
  }
  for (int s = 0; s < NT; ++s) zall[(size_t)ray * NT + s] = arr[s];
}

// ===========================================================================
// Host side
// ===========================================================================
// jax pytree leaf order (dict keys alphabetical) per net:
// color.b, color.w, density[0..7].{b,w}, density_out.{b,w}, feature.{b,w}, rgb.{b,w}
#define L_COLOR_B 0
#define L_COLOR_W 1
#define L_DENS_B(i) (2 + 2 * (i))
#define L_DENS_W(i) (3 + 2 * (i))
#define L_DO_B   18
#define L_DO_W   19
#define L_FEAT_B 20
#define L_FEAT_W 21
#define L_RGB_B  22
#define L_RGB_W  23

struct WNet { _Float16* L[8]; _Float16* feat; _Float16* color; };

static void launch_gemm(hipStream_t st, const _Float16* A, const _Float16* Wt,
                        const float* bias, _Float16* O, int M, int K,
                        int lda, int ldo, int N, int act) {
  int colG = N >> 7;                 // 2 or 1
  int rowG = 8 / colG;
  int BM = rowG << 4;                // 64 or 128
  k_gemm<<<dim3(M / BM), dim3(256), 0, st>>>(A, Wt, bias, O, K, lda, ldo, N, act);
}

static void prep_weights(hipStream_t st, const float* const* P, const WNet& Wn) {
  auto pad = [&](const float* src, _Float16* dst, int K, int Kp, int N) {
    int n = Kp * N;
    k_pad_weight_t<<<(n + 255) / 256, 256, 0, st>>>(src, dst, K, Kp, N);
  };
  pad(P[L_DENS_W(0)], Wn.L[0], 63, 64, 256);
  for (int i = 1; i < 8; ++i)
    if (i != 4) pad(P[L_DENS_W(i)], Wn.L[i], 256, 256, 256);
  pad(P[L_DENS_W(4)], Wn.L[4], 319, 320, 256);
  pad(P[L_FEAT_W], Wn.feat, 256, 256, 256);
  pad(P[L_COLOR_W], Wn.color, 283, 288, 128);
}

static void run_net(hipStream_t st, const float* const* P, const WNet& Wn,
                    const float* rays_d, int M, int nS,
                    _Float16* actA, _Float16* actB, _Float16* pin,
                    float* sigma, float* rgbp) {
  launch_gemm(st, pin,  Wn.L[0], P[L_DENS_B(0)], actA, M,  64,    64, LDACT, 256, 1);
  launch_gemm(st, actA, Wn.L[1], P[L_DENS_B(1)], actB, M, 256, LDACT, LDACT, 256, 1);
  launch_gemm(st, actB, Wn.L[2], P[L_DENS_B(2)], actA, M, 256, LDACT, LDACT, 256, 1);
  launch_gemm(st, actA, Wn.L[3], P[L_DENS_B(3)], actB, M, 256, LDACT, LDACT, 256, 1);
  launch_gemm(st, actB, Wn.L[4], P[L_DENS_B(4)], actA, M, 320, LDACT, LDACT, 256, 1); // skip concat: cols 256..319 hold pin
  launch_gemm(st, actA, Wn.L[5], P[L_DENS_B(5)], actB, M, 256, LDACT, LDACT, 256, 1);
  launch_gemm(st, actB, Wn.L[6], P[L_DENS_B(6)], actA, M, 256, LDACT, LDACT, 256, 1);
  launch_gemm(st, actA, Wn.L[7], P[L_DENS_B(7)], actB, M, 256, LDACT, LDACT, 256, 1);
  k_sigma<<<(M + 255) / 256, 256, 0, st>>>(actB, P[L_DO_W], P[L_DO_B], sigma, M);
  launch_gemm(st, actB, Wn.feat, P[L_FEAT_B], actA, M, 256, LDACT, LDACT, 256, 0);
  k_din<<<(M + 255) / 256, 256, 0, st>>>(rays_d, actA, nS, M);
  launch_gemm(st, actA, Wn.color, P[L_COLOR_B], actB, M, 288, LDACT, LDACT, 128, 1);
  k_rgb<<<(M + 255) / 256, 256, 0, st>>>(actB, P[L_RGB_W], P[L_RGB_B], rgbp, M);
}

extern "C" void kernel_launch(void* const* d_in, const int* in_sizes, int n_in,
                              void* d_out, int out_size, void* d_ws, size_t ws_size,
                              hipStream_t stream) {
  const float* rays_o = (const float*)d_in[0];
  const float* rays_d = (const float*)d_in[1];
  const float* u_c    = (const float*)d_in[2];
  const float* u_f    = (const float*)d_in[3];
  const float* const* cp = (const float* const*)&d_in[4];   // 24 coarse leaves
  const float* const* fp = (const float* const*)&d_in[28];  // 24 fine leaves
  const int* nearp = (const int*)d_in[52];
  const int* farp  = (const int*)d_in[53];
  float* out = (float*)d_out;

  const int M1 = NB * NC;      // 65536 coarse points
  const int M2 = NB * NT;      // 196608 fine points

  // --- workspace carve ---
  char* base = (char*)d_ws;
  size_t off = 0;
  auto carve = [&](size_t bytes) -> char* {
    char* p = base + off;
    off += (bytes + 255) & ~(size_t)255;
    return p;
  };
  _Float16* actA = (_Float16*)carve((size_t)M2 * LDACT * sizeof(_Float16));
  _Float16* actB = (_Float16*)carve((size_t)M2 * LDACT * sizeof(_Float16));
  _Float16* pin  = (_Float16*)carve((size_t)M2 * 64 * sizeof(_Float16));
  float* sigma = (float*)carve((size_t)M2 * sizeof(float));
  float* rgbp  = (float*)carve((size_t)M2 * 3 * sizeof(float));
  float* zc    = (float*)carve((size_t)NB * NC * sizeof(float));
  float* wc    = (float*)carve((size_t)NB * NC * sizeof(float));
  float* zall  = (float*)carve((size_t)NB * NT * sizeof(float));
  auto carve16 = [&](size_t n) { return (_Float16*)carve(n * sizeof(_Float16)); };
  WNet Wc_, Wf_;
  for (WNet* Wn : { &Wc_, &Wf_ }) {
    Wn->L[0] = carve16(64 * 256);
    for (int i = 1; i < 8; ++i) Wn->L[i] = carve16((i == 4 ? 320 : 256) * 256);
    Wn->feat  = carve16(256 * 256);
    Wn->color = carve16(288 * 128);
  }

  // --- weight conversion (f32 -> padded, transposed f16) ---
  prep_weights(stream, cp, Wc_);
  prep_weights(stream, fp, Wf_);

  // --- coarse pass ---
  k_prep_coarse<<<(M1 + 255) / 256, 256, 0, stream>>>(rays_o, rays_d, u_c, nearp, farp,
                                                      pin, actA, actB, zc);
  run_net(stream, cp, Wc_, rays_d, M1, NC, actA, actB, pin, sigma, rgbp);
  k_volrender<<<(NB + 255) / 256, 256, 0, stream>>>(sigma, rgbp, zc, out, wc, NC);

  // --- hierarchical sampling ---
  k_sample_sort<<<(NB + 255) / 256, 256, 0, stream>>>(zc, wc, u_f, zall);

  // --- fine pass ---
  k_prep_fine<<<(M2 + 255) / 256, 256, 0, stream>>>(rays_o, rays_d, zall, pin, actA, actB);
  run_net(stream, fp, Wf_, rays_d, M2, NT, actA, actB, pin, sigma, rgbp);
  k_volrender<<<(NB + 255) / 256, 256, 0, stream>>>(sigma, rgbp, zall, out + NB * 3,
                                                    nullptr, NT);
}
